// PointNet2TreeSegmentorLarge_62045097558538
// MI455X (gfx1250) — matrix operand order repaired and checked by reading, not compiled, not run on hardware
//
#include <hip/hip_runtime.h>
#include <float.h>
#include <math.h>

// ---------------------------------------------------------------------------
// PointNet++ segmentation forward for MI455X (gfx1250).
// Heavy MLP layers -> tiled WMMA GEMM (v_wmma_f32_16x16x32_f16, f32 accum),
// block tile 64x128, 8 waves, wave tile 32x32 (4 WMMAs / K-step).
// ALIGNED template variant (M%64==0, N%128==0, covers ~97% of FLOPs) has a
// completely guard-free hot loop; generic variant covers small-N layers.
// ---------------------------------------------------------------------------

typedef __attribute__((ext_vector_type(16))) _Float16 v16h;
typedef __attribute__((ext_vector_type(8)))  float    v8f;
typedef float float4a __attribute__((ext_vector_type(4))) __attribute__((aligned(4)));

#define BN_EPS 1e-5f

// ---------------------------------------------------------------------------
// Farthest point sampling: single block, deterministic start at index 0.
// ---------------------------------------------------------------------------
__global__ void fps_kernel(const float* __restrict__ pos, int n, int m,
                           int* __restrict__ idx, float* __restrict__ mind) {
  __shared__ float sv[1024];
  __shared__ int   si[1024];
  const int tid = threadIdx.x, nt = blockDim.x;
  const float p0x = pos[0], p0y = pos[1], p0z = pos[2];
  for (int i = tid; i < n; i += nt) {
    float dx = pos[3*i] - p0x, dy = pos[3*i+1] - p0y, dz = pos[3*i+2] - p0z;
    mind[i] = dx*dx + dy*dy + dz*dz;
  }
  if (tid == 0) idx[0] = 0;
  __syncthreads();
  for (int s = 1; s < m; ++s) {
    float bv = -1.f; int bi = n;
    for (int i = tid; i < n; i += nt) {
      float v = mind[i];
      if (v > bv) { bv = v; bi = i; }          // ascending scan keeps lowest index
    }
    sv[tid] = bv; si[tid] = bi;
    __syncthreads();
    for (int off = nt >> 1; off > 0; off >>= 1) {
      if (tid < off) {
        float ov = sv[tid+off]; int oi = si[tid+off];
        if (ov > sv[tid] || (ov == sv[tid] && oi < si[tid])) { sv[tid] = ov; si[tid] = oi; }
      }
      __syncthreads();
    }
    const int nxt = si[0];
    if (tid == 0) idx[s] = nxt;
    const float cx = pos[3*nxt], cy = pos[3*nxt+1], cz = pos[3*nxt+2];
    for (int i = tid; i < n; i += nt) {
      float dx = pos[3*i] - cx, dy = pos[3*i+1] - cy, dz = pos[3*i+2] - cz;
      mind[i] = fminf(mind[i], dx*dx + dy*dy + dz*dz);
    }
    __syncthreads();
  }
}

// gather centers: out[i] = src[idx[i/3]*3 + i%3]
__global__ void gather3_kernel(const float* __restrict__ src, const int* __restrict__ idx,
                               int total3, float* __restrict__ out) {
  int i = blockIdx.x * blockDim.x + threadIdx.x;
  if (i >= total3) return;
  out[i] = src[idx[i/3]*3 + (i%3)];
}

// ---------------------------------------------------------------------------
// Radius-ball k-nearest: one block per center, full distance array in LDS,
// K sequential argmin selections (tie -> lower index, matching top_k).
// ---------------------------------------------------------------------------
__global__ void radius_topk_kernel(const float* __restrict__ posc, const float* __restrict__ pos,
                                   int n, float r2, int Kn,
                                   int* __restrict__ nbr, float* __restrict__ validf) {
  __shared__ float sc[2048];
  __shared__ float rv[256];
  __shared__ int   ri[256];
  const int m = blockIdx.x, tid = threadIdx.x;
  const float cx = posc[3*m], cy = posc[3*m+1], cz = posc[3*m+2];
  for (int j = tid; j < n; j += 256) {
    float dx = pos[3*j] - cx, dy = pos[3*j+1] - cy, dz = pos[3*j+2] - cz;
    float d2 = dx*dx + dy*dy + dz*dz;
    sc[j] = (d2 <= r2) ? d2 : FLT_MAX;
  }
  __syncthreads();
  for (int s = 0; s < Kn; ++s) {
    float bv = FLT_MAX; int bi = n;
    for (int j = tid; j < n; j += 256) {
      float v = sc[j];
      if (v < bv) { bv = v; bi = j; }
    }
    rv[tid] = bv; ri[tid] = bi;
    __syncthreads();
    for (int off = 128; off > 0; off >>= 1) {
      if (tid < off) {
        float ov = rv[tid+off]; int oi = ri[tid+off];
        if (ov < rv[tid] || (ov == rv[tid] && oi < ri[tid])) { rv[tid] = ov; ri[tid] = oi; }
      }
      __syncthreads();
    }
    if (tid == 0) {
      if (rv[0] < FLT_MAX) { nbr[m*Kn+s] = ri[0]; validf[m*Kn+s] = 1.f; sc[ri[0]] = FLT_MAX; }
      else                 { nbr[m*Kn+s] = 0;     validf[m*Kn+s] = 0.f; }
    }
    __syncthreads();
  }
}

// Build grouped PointNetConv input rows: [x[nbr], pos[nbr]-pos_c]; zeros if invalid.
__global__ void gather_feats_kernel(const float* __restrict__ X, int C,
                                    const float* __restrict__ pos, const float* __restrict__ posc,
                                    const int* __restrict__ nbr, const float* __restrict__ validf,
                                    int Kn, float* __restrict__ feats) {
  const int row = blockIdx.x;          // m*Kn + k
  const int m = row / Kn;
  const int j = nbr[row];
  const float v = validf[row];
  const int Cin = C + 3;
  for (int c = threadIdx.x; c < Cin; c += blockDim.x) {
    float val = 0.f;
    if (v > 0.5f) {
      if (c < C) val = X[(size_t)j*C + c];
      else       val = pos[3*j + (c - C)] - posc[3*m + (c - C)];
    }
    feats[(size_t)row*Cin + c] = val;
  }
}

// ---------------------------------------------------------------------------
// WMMA GEMM: C[M,N] = A[M,K] @ W[K,N] (+bias).
// ---------------------------------------------------------------------------
#define GM 64
#define GN 128
#define LDAH 40   // LDS half stride (even, dword-aligned pairs, pads banks)

union Frag { v16h v; unsigned int u[8]; _Float16 h[16]; };

__device__ __forceinline__ v8f do_wmma(const Frag& a, const Frag& b, v8f acc) {
  return __builtin_amdgcn_wmma_f32_16x16x32_f16(
      /*neg_a=*/false, a.v, /*neg_b=*/false, b.v,
      /*c_mod=*/(short)0, acc, /*reuse_a=*/false, /*reuse_b=*/false);
}

// Stage one 64x32 A tile and 32x128 W tile into LDS as f16.
// ALIGNED: no row/col guards. FULLK: no K guards.
template <bool ALIGNED, bool FULLK>
__device__ __forceinline__ void stage_tiles(
    const float* __restrict__ A, const float* __restrict__ W,
    _Float16* __restrict__ As, _Float16* __restrict__ Bs,
    int tid, int rowBase, int colBase, int k0, int M, int Kdim, int N) {
  // A tile: float4 groups along K
  for (int g = tid; g < (GM * 32) / 4; g += 256) {
    int r = g >> 3, k4 = (g & 7) << 2;
    int gr = rowBase + r, gk = k0 + k4;
    _Float16* dst = &As[r * LDAH + k4];
    const bool rowOK = ALIGNED || (gr < M);
    const bool kOK4  = FULLK   || (gk + 4 <= Kdim);
    if (rowOK && kOK4) {
      float4a v = *(const float4a*)(A + (size_t)gr * Kdim + gk);
      dst[0] = (_Float16)v.x; dst[1] = (_Float16)v.y;
      dst[2] = (_Float16)v.z; dst[3] = (_Float16)v.w;
    } else {
#pragma unroll
      for (int i = 0; i < 4; ++i) {
        float v = (rowOK && (FULLK || gk + i < Kdim)) ? A[(size_t)gr * Kdim + gk + i] : 0.f;
        dst[i] = (_Float16)v;
      }
    }
  }
  // W tile: float4 groups along N, stored col(N)-major (K contiguous)
  for (int g = tid; g < (32 * GN) / 4; g += 256) {
    int r = g >> 5, c4 = (g & 31) << 2;
    int gk = k0 + r, gc = colBase + c4;
    const bool kOK   = FULLK   || (gk < Kdim);
    const bool colOK = ALIGNED || (gc + 4 <= N);
    if (kOK && colOK) {
      float4a v = *(const float4a*)(W + (size_t)gk * N + gc);
      Bs[(c4 + 0) * LDAH + r] = (_Float16)v.x;
      Bs[(c4 + 1) * LDAH + r] = (_Float16)v.y;
      Bs[(c4 + 2) * LDAH + r] = (_Float16)v.z;
      Bs[(c4 + 3) * LDAH + r] = (_Float16)v.w;
    } else {
#pragma unroll
      for (int i = 0; i < 4; ++i) {
        float v = (kOK && (ALIGNED || gc + i < N)) ? W[(size_t)gk * N + gc + i] : 0.f;
        Bs[(c4 + i) * LDAH + r] = (_Float16)v;
      }
    }
  }
}

// Fragment loads + 4 WMMAs for one 32-K step.
__device__ __forceinline__ void compute_step(
    const _Float16* __restrict__ As, const _Float16* __restrict__ Bs,
    int wr, int wc, int half, int l16,
    v8f& acc00, v8f& acc01, v8f& acc10, v8f& acc11) {
  Frag a0, a1, b0, b1;
  // A: row = base + l16; VGPR v holds K-pair p = (v<4?v:v+4) + 4*half
  const _Float16* Ar0 = &As[(wr * 32 + l16) * LDAH];
  const _Float16* Ar1 = &As[(wr * 32 + 16 + l16) * LDAH];
#pragma unroll
  for (int v = 0; v < 8; ++v) {
    int p = (v < 4 ? v : v + 4) + 4 * half;
    a0.u[v] = *(const unsigned int*)&Ar0[2 * p];
    a1.u[v] = *(const unsigned int*)&Ar1[2 * p];
  }
  // B: col = base + l16; VGPR v holds K-pair p = (v<4?v:v+4) + 4*half
  const _Float16* Bc0 = &Bs[(wc * 32 + l16) * LDAH];
  const _Float16* Bc1 = &Bs[(wc * 32 + 16 + l16) * LDAH];
#pragma unroll
  for (int v = 0; v < 8; ++v) {
    b0.u[v] = *(const unsigned int*)&Bc0[2 * (v < 4 ? v : v + 4) + 8 * half];
    b1.u[v] = *(const unsigned int*)&Bc1[2 * (v < 4 ? v : v + 4) + 8 * half];
  }
  acc00 = do_wmma(a0, b0, acc00);
  acc01 = do_wmma(a0, b1, acc01);
  acc10 = do_wmma(a1, b0, acc10);
  acc11 = do_wmma(a1, b1, acc11);
}

template <bool ALIGNED>
__global__ __launch_bounds__(256) void wmma_gemm_kernel(
    const float* __restrict__ A, const float* __restrict__ W,
    const float* __restrict__ bias, float* __restrict__ C,
    int M, int Kdim, int N) {
  __shared__ _Float16 As[GM * LDAH];
  __shared__ _Float16 Bs[GN * LDAH];
  const int tid  = threadIdx.x;
  const int lane = tid & 31;
  const int wave = tid >> 5;           // 0..7
  const int wr   = wave >> 2;          // 0..1 : 32-row sub-tile
  const int wc   = wave & 3;           // 0..3 : 32-col sub-tile
  const int rowBase = blockIdx.y * GM;
  const int colBase = blockIdx.x * GN;
  const int half = lane >> 4;
  const int l16  = lane & 15;

  v8f acc00 = {}, acc01 = {}, acc10 = {}, acc11 = {};

  const int Kfull = Kdim & ~31;
  int k0 = 0;
  for (; k0 < Kfull; k0 += 32) {        // guard-free hot loop (when ALIGNED)
    stage_tiles<ALIGNED, true>(A, W, As, Bs, tid, rowBase, colBase, k0, M, Kdim, N);
    __syncthreads();
    compute_step(As, Bs, wr, wc, half, l16, acc00, acc01, acc10, acc11);
    __syncthreads();
  }
  if (k0 < Kdim) {                      // K tail (zero-filled)
    stage_tiles<ALIGNED, false>(A, W, As, Bs, tid, rowBase, colBase, k0, M, Kdim, N);
    __syncthreads();
    compute_step(As, Bs, wr, wc, half, l16, acc00, acc01, acc10, acc11);
    __syncthreads();
  }

  // store: C/D layout VGPR r -> (M = r + 8*half, N = l16)
  auto store_tile = [&](const v8f& acc, int r0, int c0) {
    int col = c0 + l16;
    if (ALIGNED || col < N) {
      float bc = bias ? bias[col] : 0.f;
#pragma unroll
      for (int r = 0; r < 8; ++r) {
        int row = r0 + r + 8 * half;
        if (ALIGNED || row < M) C[(size_t)row * N + col] = acc[r] + bc;
      }
    }
  };
  const int r0 = rowBase + wr * 32;
  const int c0 = colBase + wc * 32;
  store_tile(acc00, r0,      c0);
  store_tile(acc01, r0,      c0 + 16);
  store_tile(acc10, r0 + 16, c0);
  store_tile(acc11, r0 + 16, c0 + 16);
}

// ---------------------------------------------------------------------------
// BatchNorm1d training-mode stats (masked, biased var) + apply + ReLU.
// ---------------------------------------------------------------------------
__global__ void bn_stats_kernel(const float* __restrict__ Y, const float* __restrict__ maskRow,
                                int M, int N, float* __restrict__ mean, float* __restrict__ var) {
  __shared__ float s0[256], s1[256], s2[256];
  const int c = blockIdx.x, tid = threadIdx.x;
  float a = 0.f, b = 0.f, w = 0.f;
  for (int r = tid; r < M; r += 256) {
    float wv = maskRow ? maskRow[r] : 1.f;
    float v  = Y[(size_t)r * N + c];
    a += v * wv; b += v * v * wv; w += wv;
  }
  s0[tid] = a; s1[tid] = b; s2[tid] = w;
  __syncthreads();
  for (int off = 128; off > 0; off >>= 1) {
    if (tid < off) { s0[tid] += s0[tid+off]; s1[tid] += s1[tid+off]; s2[tid] += s2[tid+off]; }
    __syncthreads();
  }
  if (tid == 0) {
    float cnt = fmaxf(s2[0], 1.f);
    float mu  = s0[0] / cnt;
    float vv  = s1[0] / cnt - mu * mu;
    mean[c] = mu; var[c] = fmaxf(vv, 0.f);
  }
}

__global__ void bn_relu_kernel(float* __restrict__ Y, const float* __restrict__ mean,
                               const float* __restrict__ var, const float* __restrict__ gamma,
                               const float* __restrict__ beta, size_t total, int N) {
  size_t i = (size_t)blockIdx.x * blockDim.x + threadIdx.x;
  if (i >= total) return;
  int c = (int)(i % (size_t)N);
  float v = Y[i];
  v = gamma[c] * (v - mean[c]) * rsqrtf(var[c] + BN_EPS) + beta[c];
  Y[i] = v > 0.f ? v : 0.f;
}

// Masked max aggregation over the K neighbors of each center.
__global__ void max_agg_kernel(const float* __restrict__ msg, const float* __restrict__ validf,
                               int Kn, int C, float* __restrict__ out) {
  const int m = blockIdx.x;
  for (int c = threadIdx.x; c < C; c += blockDim.x) {
    float b = -FLT_MAX;
    for (int k = 0; k < Kn; ++k)
      if (validf[m*Kn + k] > 0.5f) b = fmaxf(b, msg[((size_t)m*Kn + k)*C + c]);
    out[(size_t)m*C + c] = b;
  }
}

// ---------------------------------------------------------------------------
// knn_interpolate (k=3): one block per target point.
// ---------------------------------------------------------------------------
__global__ void knn3_kernel(const float* __restrict__ posy, const float* __restrict__ posx,
                            int Nx, const float* __restrict__ X, int C,
                            float* __restrict__ out) {
  __shared__ float sd[1024];
  __shared__ float rv[256];
  __shared__ int   ri[256];
  __shared__ float sw[3];
  __shared__ int   sid[3];
  const int y = blockIdx.x, tid = threadIdx.x;
  const float cx = posy[3*y], cy = posy[3*y+1], cz = posy[3*y+2];
  for (int j = tid; j < Nx; j += 256) {
    float dx = posx[3*j] - cx, dy = posx[3*j+1] - cy, dz = posx[3*j+2] - cz;
    sd[j] = dx*dx + dy*dy + dz*dz;
  }
  __syncthreads();
  for (int t = 0; t < 3; ++t) {
    float bv = FLT_MAX; int bi = Nx;
    for (int j = tid; j < Nx; j += 256) {
      float v = sd[j];
      if (v < bv) { bv = v; bi = j; }
    }
    rv[tid] = bv; ri[tid] = bi;
    __syncthreads();
    for (int off = 128; off > 0; off >>= 1) {
      if (tid < off) {
        float ov = rv[tid+off]; int oi = ri[tid+off];
        if (ov < rv[tid] || (ov == rv[tid] && oi < ri[tid])) { rv[tid] = ov; ri[tid] = oi; }
      }
      __syncthreads();
    }
    if (tid == 0) { sid[t] = ri[0]; sw[t] = 1.f / fmaxf(rv[0], 1e-16f); sd[ri[0]] = FLT_MAX; }
    __syncthreads();
  }
  const float w0 = sw[0], w1 = sw[1], w2 = sw[2];
  const float wsum = w0 + w1 + w2;
  const int i0 = sid[0], i1 = sid[1], i2 = sid[2];
  for (int c = tid; c < C; c += 256)
    out[(size_t)y*C + c] =
        (w0 * X[(size_t)i0*C + c] + w1 * X[(size_t)i1*C + c] + w2 * X[(size_t)i2*C + c]) / wsum;
}

// concat(A, B, zeros[nz]) row-wise
__global__ void concat_kernel(const float* __restrict__ A, int Ca,
                              const float* __restrict__ B, int Cb, int nz,
                              float* __restrict__ out) {
  const int row = blockIdx.x;
  const int Cout = Ca + Cb + nz;
  for (int c = threadIdx.x; c < Cout; c += blockDim.x) {
    float v = 0.f;
    if (c < Ca)           v = A[(size_t)row*Ca + c];
    else if (c < Ca + Cb) v = B[(size_t)row*Cb + (c - Ca)];
    out[(size_t)row*Cout + c] = v;
  }
}

// ---------------------------------------------------------------------------
// Host orchestration
// ---------------------------------------------------------------------------
extern "C" void kernel_launch(void* const* d_in, const int* in_sizes, int n_in,
                              void* d_out, int out_size, void* d_ws, size_t ws_size,
                              hipStream_t stream) {
  if (n_in < 60) return;
  auto F = [&](int i) { return (const float*)d_in[i]; };

  // ---- parameters (x, pos, then params in JAX pytree order) ----
  const float* x   = F(0);
  const float* pos = F(1);
  // reg: [67,32,16,1] no-bias
  const float *regW0=F(2),  *regBe0=F(3),  *regGa0=F(4);
  const float *regW1=F(5),  *regBe1=F(6),  *regGa1=F(7);
  const float *regW2=F(8);
  // sa0: [6,64,128,256] bias
  const float *s0W0=F(9),  *s0b0=F(10), *s0Be0=F(11), *s0Ga0=F(12);
  const float *s0W1=F(13), *s0b1=F(14), *s0Be1=F(15), *s0Ga1=F(16);
  const float *s0W2=F(17), *s0b2=F(18);
  // sa1: [259,512,1024,2048]
  const float *s1W0=F(19), *s1b0=F(20), *s1Be0=F(21), *s1Ga0=F(22);
  const float *s1W1=F(23), *s1b1=F(24), *s1Be1=F(25), *s1Ga1=F(26);
  const float *s1W2=F(27), *s1b2=F(28);
  // sa2: [2051,2048,2048,2048]
  const float *s2W0=F(29), *s2b0=F(30), *s2Be0=F(31), *s2Ga0=F(32);
  const float *s2W1=F(33), *s2b1=F(34), *s2Be1=F(35), *s2Ga1=F(36);
  const float *s2W2=F(37), *s2b2=F(38);
  // upn0: [4099,2048,1024,512] no-bias
  const float *u0W0=F(39), *u0Be0=F(40), *u0Ga0=F(41);
  const float *u0W1=F(42), *u0Be1=F(43), *u0Ga1=F(44);
  const float *u0W2=F(45);
  // upn1: [771,2048,1024,512]
  const float *u1W0=F(46), *u1Be0=F(47), *u1Ga0=F(48);
  const float *u1W1=F(49), *u1Be1=F(50), *u1Ga1=F(51);
  const float *u1W2=F(52);
  // upn2: [518,256,128,64]
  const float *u2W0=F(53), *u2Be0=F(54), *u2Ga0=F(55);
  const float *u2W1=F(56), *u2Be1=F(57), *u2Ga1=F(58);
  const float *u2W2=F(59);

  // ---- workspace carve-up ----
  char* wsb = (char*)d_ws;
  size_t off = 0;
  auto alloc = [&](size_t nbytes) -> void* {
    void* p = wsb + off;
    off = (off + nbytes + 255) & ~(size_t)255;
    return p;
  };
  float* mind   = (float*)alloc(2048 * 4);
  int*   idx0   = (int*)  alloc(1024 * 4);
  int*   idx1   = (int*)  alloc(512 * 4);
  int*   idx2   = (int*)  alloc(256 * 4);
  float* pos0   = (float*)alloc(1024 * 3 * 4);
  float* pos1   = (float*)alloc(512 * 3 * 4);
  float* pos2   = (float*)alloc(256 * 3 * 4);
  int*   nbr    = (int*)  alloc(32768 * 4);
  float* validf = (float*)alloc(32768 * 4);
  float* meanP  = (float*)alloc(2048 * 4);
  float* varP   = (float*)alloc(2048 * 4);
  float* x0     = (float*)alloc((size_t)1024 * 256 * 4);
  float* x1     = (float*)alloc((size_t)512 * 2048 * 4);
  float* x2     = (float*)alloc((size_t)256 * 2048 * 4);
  const size_t BUF_ELEMS = (size_t)16384 * 2048;   // largest activation
  float* bufA = (float*)alloc(BUF_ELEMS * 4);
  float* bufB = (float*)alloc(BUF_ELEMS * 4);
  (void)ws_size; (void)in_sizes; (void)out_size;

  // ---- helpers ----
  auto gemm = [&](const float* A, const float* W, const float* b, float* C,
                  int M, int K, int N) {
    dim3 grid((N + GN - 1) / GN, (M + GM - 1) / GM);
    if ((M % GM) == 0 && (N % GN) == 0)
      wmma_gemm_kernel<true><<<grid, 256, 0, stream>>>(A, W, b, C, M, K, N);
    else
      wmma_gemm_kernel<false><<<grid, 256, 0, stream>>>(A, W, b, C, M, K, N);
  };
  auto bnrelu = [&](float* Y, const float* maskRow, int M, int N,
                    const float* gamma, const float* beta) {
    bn_stats_kernel<<<N, 256, 0, stream>>>(Y, maskRow, M, N, meanP, varP);
    size_t total = (size_t)M * N;
    bn_relu_kernel<<<(unsigned)((total + 255) / 256), 256, 0, stream>>>(
        Y, meanP, varP, gamma, beta, total, N);
  };
  auto set_abstraction = [&](const float* Xin, int Cx, const float* posIn, int n, int m,
                             int* idxP, float* posC,
                             const float* W0, const float* b0, const float* Be0, const float* Ga0, int C1,
                             const float* W1, const float* b1, const float* Be1, const float* Ga1, int C2,
                             const float* W2, const float* b2, int C3,
                             float* xOut) {
    fps_kernel<<<1, 1024, 0, stream>>>(posIn, n, m, idxP, mind);
    gather3_kernel<<<(m * 3 + 255) / 256, 256, 0, stream>>>(posIn, idxP, m * 3, posC);
    radius_topk_kernel<<<m, 256, 0, stream>>>(posC, posIn, n, 0.16f /*r^2*/, 32, nbr, validf);
    const int rows = m * 32, Cin = Cx + 3;
    gather_feats_kernel<<<rows, 256, 0, stream>>>(Xin, Cx, posIn, posC, nbr, validf, 32, bufA);
    gemm(bufA, W0, b0, bufB, rows, Cin, C1); bnrelu(bufB, validf, rows, C1, Ga0, Be0);
    gemm(bufB, W1, b1, bufA, rows, C1, C2);  bnrelu(bufA, validf, rows, C2, Ga1, Be1);
    gemm(bufA, W2, b2, bufB, rows, C2, C3);
    max_agg_kernel<<<m, 256, 0, stream>>>(bufB, validf, 32, C3, xOut);
  };

  // ---- set abstraction pyramid ----
  set_abstraction(x,  3,    pos,  2048, 1024, idx0, pos0,
                  s0W0, s0b0, s0Be0, s0Ga0, 64,
                  s0W1, s0b1, s0Be1, s0Ga1, 128,
                  s0W2, s0b2, 256, x0);
  set_abstraction(x0, 256,  pos0, 1024, 512,  idx1, pos1,
                  s1W0, s1b0, s1Be0, s1Ga0, 512,
                  s1W1, s1b1, s1Be1, s1Ga1, 1024,
                  s1W2, s1b2, 2048, x1);
  set_abstraction(x1, 2048, pos1, 512,  256,  idx2, pos2,
                  s2W0, s2b0, s2Be0, s2Ga0, 2048,
                  s2W1, s2b1, s2Be1, s2Ga1, 2048,
                  s2W2, s2b2, 2048, x2);

  // ---- FP level 0: interp(x2: pos2 -> pos1), concat x1, upn0 ----
  knn3_kernel<<<512, 256, 0, stream>>>(pos1, pos2, 256, x2, 2048, bufA);
  concat_kernel<<<512, 256, 0, stream>>>(bufA, 2048, x1, 2048, 3, bufB);   // [512,4099]
  gemm(bufB, u0W0, nullptr, bufA, 512, 4099, 2048); bnrelu(bufA, nullptr, 512, 2048, u0Ga0, u0Be0);
  gemm(bufA, u0W1, nullptr, bufB, 512, 2048, 1024); bnrelu(bufB, nullptr, 512, 1024, u0Ga1, u0Be1);
  gemm(bufB, u0W2, nullptr, bufA, 512, 1024, 512);                          // h0 = bufA [512,512]

  // ---- FP level 1 ----
  knn3_kernel<<<1024, 256, 0, stream>>>(pos0, pos1, 512, bufA, 512, bufB);  // [1024,512]
  concat_kernel<<<1024, 256, 0, stream>>>(bufB, 512, x0, 256, 3, bufA);     // [1024,771]
  gemm(bufA, u1W0, nullptr, bufB, 1024, 771, 2048); bnrelu(bufB, nullptr, 1024, 2048, u1Ga0, u1Be0);
  gemm(bufB, u1W1, nullptr, bufA, 1024, 2048, 1024); bnrelu(bufA, nullptr, 1024, 1024, u1Ga1, u1Be1);
  gemm(bufA, u1W2, nullptr, bufB, 1024, 1024, 512);                         // h1 = bufB [1024,512]

  // ---- FP level 2 ----
  knn3_kernel<<<2048, 256, 0, stream>>>(pos, pos0, 1024, bufB, 512, bufA);  // [2048,512]
  concat_kernel<<<2048, 256, 0, stream>>>(bufA, 512, x, 3, 3, bufB);        // [2048,518]
  gemm(bufB, u2W0, nullptr, bufA, 2048, 518, 256); bnrelu(bufA, nullptr, 2048, 256, u2Ga0, u2Be0);
  gemm(bufA, u2W1, nullptr, bufB, 2048, 256, 128); bnrelu(bufB, nullptr, 2048, 128, u2Ga1, u2Be1);
  gemm(bufB, u2W2, nullptr, bufA, 2048, 128, 64);                           // h2 = bufA [2048,64]

  // ---- regression head: concat(h2, pos) -> [2048,67] -> 32 -> 16 -> 1 ----
  concat_kernel<<<2048, 256, 0, stream>>>(bufA, 64, pos, 3, 0, bufB);       // [2048,67]
  gemm(bufB, regW0, nullptr, bufA, 2048, 67, 32); bnrelu(bufA, nullptr, 2048, 32, regGa0, regBe0);
  gemm(bufA, regW1, nullptr, bufB, 2048, 32, 16); bnrelu(bufB, nullptr, 2048, 16, regGa1, regBe1);
  gemm(bufB, regW2, nullptr, (float*)d_out, 2048, 16, 1);                   // [2048,1]
}
